// SparseBEVTransformerDecoder_3006477107634
// MI455X (gfx1250) — matrix operand order repaired and checked by reading, not compiled, not run on hardware
//
#include <hip/hip_runtime.h>
#include <stdint.h>

// ---------------------------------------------------------------------------
// Types for CDNA5 WMMA (wave32, 16x16x32 bf16 -> f32)
// ---------------------------------------------------------------------------
typedef __attribute__((ext_vector_type(16))) __bf16          v16bf;
typedef __attribute__((ext_vector_type(16))) unsigned short  v16us;
typedef __attribute__((ext_vector_type(8)))  float           v8f;
typedef __attribute__((ext_vector_type(4)))  int             v4i;

union Frag {
    v16us u;
    v16bf b;
};

#define Q_TOT 40000
#define EMB   128
#define NP    8
#define NCAM  6

// Use gfx1250 async global->LDS copies in the GEMM (set to 0 to fall back to
// VGPR-staged ds_store path).
#define USE_ASYNC_LDS 1

// Native f32 -> bf16 (v_cvt_pk_bf16_f32) instead of manual RNE bit twiddling.
__device__ __forceinline__ unsigned short f2bf(float f) {
    __bf16 h = (__bf16)f;
    return __builtin_bit_cast(unsigned short, h);
}
__device__ __forceinline__ float bf2f(unsigned short h) {
    return (float)__builtin_bit_cast(__bf16, h);
}

__device__ __forceinline__ void wait_asynccnt0() {
#if __has_builtin(__builtin_amdgcn_s_wait_asynccnt)
    __builtin_amdgcn_s_wait_asynccnt(0);
#else
    asm volatile("s_wait_asynccnt 0x0" ::: "memory");
#endif
}

// ---------------------------------------------------------------------------
// Repack fp32 weight [K x Nf] row-major -> bf16 B-fragment layout:
//   Bpk[(kstep*(Nf/16)+ntile)*32 + lane][j],  k = kstep*32 + j + 8*((j>=8)+(lane>=16)),
//   n = ntile*16 + lane%16   (ISA 7.12.2 16-bit B-matrix layout)
// ---------------------------------------------------------------------------
__global__ void repack_w(const float* __restrict__ W, unsigned short* __restrict__ Pk,
                         int K, int Nf) {
    int tid = blockIdx.x * blockDim.x + threadIdx.x;
    int total = (K / 32) * (Nf / 16) * 32;
    if (tid >= total) return;
    int lane = tid & 31;
    int nt   = (tid >> 5) % (Nf / 16);
    int ks   = tid / (32 * (Nf / 16));
    int n     = nt * 16 + (lane & 15);
    int kbase = ks * 32;
    Frag o;
#pragma unroll
    for (int j = 0; j < 16; ++j) o.u[j] = f2bf(W[(size_t)(kbase + j + 8 * ((j >= 8) + (lane >> 4))) * Nf + n]);
    *(v16us*)(Pk + (size_t)tid * 16) = o.u;
}

// ---------------------------------------------------------------------------
// Sampling + projection + scale softmax + fused PE MLP (layer2 on WMMA),
// output = feat tile packed in A-fragment layout (bf16), 16 queries / block.
//   A-fragment layout (16-bit A 16x32): lane holds row M=lane%16,
//   k = kstep*32 + j + 8*((j>=8)+(lane>=16))
// ---------------------------------------------------------------------------
__global__ __launch_bounds__(256)
void sample_kernel(const float* __restrict__ f0, const float* __restrict__ f1,
                   const float* __restrict__ f2, const float* __restrict__ f3,
                   const float* __restrict__ query, const float* __restrict__ query_pos,
                   const float* __restrict__ l2i, const float* __restrict__ Wsw,
                   const float* __restrict__ bsw,
                   const float* __restrict__ pew1, const float* __restrict__ peb1,
                   const unsigned short* __restrict__ pw2pk, const float* __restrict__ peb2,
                   unsigned short* __restrict__ featPk) {
    __shared__ unsigned short s_feat[16 * 1024];   // 32 KB, bf16 feat tile [q][p*128+g*32+c]
    __shared__ float s_sw[128 * 16];               // 8 KB softmaxed scale weights
    __shared__ float s_qp[128 * 3];
    __shared__ float s_u[128], s_v[128];
    __shared__ int   s_cam[128], s_ok[128];
    __shared__ float s_w1[3 * 256];
    __shared__ float s_b1[256];
    __shared__ float s_m[96];

    const int t    = threadIdx.x;
    const int qblk = blockIdx.x;                   // 0..2499, rowtile of 16 queries

    for (int i = t; i < 768; i += 256) s_w1[i] = pew1[i];
    if (t < 256) s_b1[t] = peb1[t];
    if (t < 96)  s_m[t]  = l2i[t];
    if (t < 128) {
        int p = t & 7, ql = t >> 3;
        int q = qblk * 16 + ql;
        for (int c = 0; c < 3; ++c)
            s_qp[t * 3 + c] = query_pos[((size_t)p * Q_TOT + q) * 3 + c];
    }
    __syncthreads();

    // ---- per (query, point): projection + first-valid camera + scale softmax
    if (t < 128) {
        int p = t & 7, ql = t >> 3;
        int q = qblk * 16 + ql;
        float px = s_qp[t * 3 + 0] * 100.f - 50.f;
        float py = s_qp[t * 3 + 1] * 100.f - 50.f;
        float pz = s_qp[t * 3 + 2] * 8.f - 4.f;
        int cam = 0, ok = 0;
        float us = 0.f, vs = 0.f;
        for (int n = 0; n < NCAM; ++n) {
            const float* M = &s_m[n * 16];
            float c0 = M[0] * px + M[1] * py + M[2] * pz + M[3];
            float c1 = M[4] * px + M[5] * py + M[6] * pz + M[7];
            float c2 = M[8] * px + M[9] * py + M[10] * pz + M[11];
            float d  = fmaxf(c2, 1e-6f);
            float u  = (c0 / d) * (1.f / 480.f);
            float v  = (c1 / d) * (1.f / 224.f);
            int val = (c2 > 1e-6f) & (u > 0.f) & (u < 1.f) & (v > 0.f) & (v < 1.f);
            if (val && !ok) { ok = 1; cam = n; us = u; vs = v; }
        }
        s_u[t] = us; s_v[t] = vs; s_cam[t] = cam; s_ok[t] = ok;

        const float* qr = query + ((size_t)q * NP + p) * EMB;
        float lg[16];
#pragma unroll
        for (int j = 0; j < 16; ++j) lg[j] = bsw[j];
        for (int e = 0; e < EMB; ++e) {
            float qv = qr[e];
            const float* wr = Wsw + e * 16;
#pragma unroll
            for (int j = 0; j < 16; ++j) lg[j] += qv * wr[j];
        }
#pragma unroll
        for (int g = 0; g < 4; ++g) {
            float a0 = lg[g * 4 + 0], a1 = lg[g * 4 + 1];
            float a2 = lg[g * 4 + 2], a3 = lg[g * 4 + 3];
            float mx = fmaxf(fmaxf(a0, a1), fmaxf(a2, a3));
            float e0 = __expf(a0 - mx), e1 = __expf(a1 - mx);
            float e2 = __expf(a2 - mx), e3 = __expf(a3 - mx);
            float inv = 1.f / (e0 + e1 + e2 + e3);
            s_sw[t * 16 + g * 4 + 0] = e0 * inv;
            s_sw[t * 16 + g * 4 + 1] = e1 * inv;
            s_sw[t * 16 + g * 4 + 2] = e2 * inv;
            s_sw[t * 16 + g * 4 + 3] = e3 * inv;
        }
    }
    __syncthreads();

    // ---- multi-level bilinear gather (feature levels live in L2)
    {
        const float* fp[4] = {f0, f1, f2, f3};
        const int HH[4] = {56, 28, 14, 7};
        const int WW[4] = {120, 60, 30, 15};
        int c = t & 31;
        for (int qgp = t >> 5; qgp < 512; qgp += 8) {
            int ql = qgp >> 5, g = (qgp >> 3) & 3, p = qgp & 7;
            int idx = ql * 8 + p;
            float acc = 0.f;
            if (s_ok[idx]) {
                int cam = s_cam[idx];
                float u = s_u[idx], v = s_v[idx];
#pragma unroll
                for (int l = 0; l < 4; ++l) {
                    int H = HH[l], W = WW[l];
                    float x = u * W - 0.5f, y = v * H - 0.5f;
                    float x0f = floorf(x), y0f = floorf(y);
                    float wx = x - x0f, wy = y - y0f;
                    int x0 = (int)x0f, y0 = (int)y0f;
                    int x1 = x0 + 1, y1 = y0 + 1;
                    const float* base = fp[l] + ((size_t)cam * 128 + g * 32 + c) * (size_t)H * W;
                    float s00 = 0.f, s01 = 0.f, s10 = 0.f, s11 = 0.f;
                    bool xi0 = (x0 >= 0) & (x0 < W), xi1 = (x1 >= 0) & (x1 < W);
                    bool yi0 = (y0 >= 0) & (y0 < H), yi1 = (y1 >= 0) & (y1 < H);
                    if (yi0 & xi0) s00 = base[y0 * W + x0];
                    if (yi0 & xi1) s01 = base[y0 * W + x1];
                    if (yi1 & xi0) s10 = base[y1 * W + x0];
                    if (yi1 & xi1) s11 = base[y1 * W + x1];
                    float bil = s00 * (1.f - wx) * (1.f - wy) + s01 * wx * (1.f - wy)
                              + s10 * (1.f - wx) * wy + s11 * wx * wy;
                    acc += bil * s_sw[idx * 16 + g * 4 + l];
                }
            }
            s_feat[ql * 1024 + p * 128 + g * 32 + c] = f2bf(acc);
        }
    }
    __syncthreads();

    // ---- fused PE MLP: layer1 (K=3, inline per lane) + layer2 on WMMA.
    //      128 local rows = 8 M-tiles, one per wave; Nf = 128 = 8 N-tiles.
    {
        int wave = t >> 5, lane = t & 31, ln = lane & 15, hi = lane >> 4;
        int rowl = wave * 16 + ln;
        float q0 = s_qp[rowl * 3 + 0], q1 = s_qp[rowl * 3 + 1], q2 = s_qp[rowl * 3 + 2];
        Frag af[8];
#pragma unroll
        for (int ks = 0; ks < 8; ++ks) {
#pragma unroll
            for (int j = 0; j < 16; ++j) {
                int k = ks * 32 + j + 8 * ((j >= 8) + hi);
                float h = s_b1[k] + q0 * s_w1[k] + q1 * s_w1[256 + k] + q2 * s_w1[512 + k];
                af[ks].u[j] = f2bf(fmaxf(h, 0.f));
            }
        }
#pragma unroll
        for (int nt = 0; nt < 8; ++nt) {
            v8f acc = {0.f, 0.f, 0.f, 0.f, 0.f, 0.f, 0.f, 0.f};
#pragma unroll
            for (int ks = 0; ks < 8; ++ks) {
                Frag bf_;
                bf_.u = *(const v16us*)(pw2pk + ((size_t)(ks * 8 + nt) * 32 + lane) * 16);
                acc = __builtin_amdgcn_wmma_f32_16x16x32_bf16(
                    false, af[ks].b, false, bf_.b, (short)0, acc, false, false);
            }
            int n = nt * 16 + ln;
            float b2 = peb2[n];
#pragma unroll
            for (int r = 0; r < 8; ++r) {
                int m = wave * 16 + r + 8 * hi;            // local row = q_local*8 + p
                int si = (m >> 3) * 1024 + (m & 7) * 128 + n;
                s_feat[si] = f2bf(bf2f(s_feat[si]) + acc[r] + b2);
            }
        }
    }
    __syncthreads();

    // ---- pack 16x1024 tile to A-fragment layout (rowtile = qblk, ksteps = 32)
    for (int s = 0; s < 4; ++s) {
        int slot = t + s * 256;
        int ks = slot >> 5, lane2 = slot & 31;
        int row = lane2 & 15, hi2 = lane2 >> 4;
        Frag o;
#pragma unroll
        for (int j = 0; j < 16; ++j) {
            int k = ks * 32 + j + 8 * ((j >= 8) + hi2);
            o.u[j] = s_feat[row * 1024 + k];
        }
        *(v16us*)(featPk + ((size_t)(qblk * 32 + ks) * 32 + lane2) * 16) = o.u;
    }
}

// ---------------------------------------------------------------------------
// WMMA GEMM: out = act(A[rows x K] @ W[K x Nf] + bias)
//   A in A-fragment bf16 layout, W in B-fragment bf16 layout.
//   Block: 8 waves -> 128 rows x 64 cols.
//   B fragments staged in LDS with a double-buffered software pipeline built
//   on GLOBAL_LOAD_ASYNC_TO_LDS_B128 (ASYNCcnt), overlapping copy with WMMA.
//   OUT_MODE 0: bf16 A-fragment layout (feeds next GEMM)
//   OUT_MODE 2: fp32 transposed final store  out[n*rows + m]
// ---------------------------------------------------------------------------
template <int OUT_MODE, bool RELU>
__global__ __launch_bounds__(256)
void gemm_kernel(const unsigned short* __restrict__ A, const unsigned short* __restrict__ B,
                 const float* __restrict__ bias, void* __restrict__ outp,
                 int rows, int K, int Nf) {
    __shared__ unsigned short ldsB[2][2048];       // 2 x 4KB: 4 ntiles x 32 lanes x 16
    const int t = threadIdx.x;
    const int wave = t >> 5, lane = t & 31, ln = lane & 15, hi = lane >> 4;
    const int ksteps = K >> 5;
    const int RT = (rows + 15) >> 4;
    const int rowtile = blockIdx.y * 8 + wave;
    const int nt0 = blockIdx.x * 4;
    const int nTiles = Nf >> 4;
    const bool act = rowtile < RT;

    // per-thread source chunk (16 B) within a 4 KB B tile
    const int ntl    = t >> 6;
    const int lane_s = (t & 63) >> 1;
    const int j0     = (t & 1) << 3;

#if USE_ASYNC_LDS
    auto issueB = [&](int ks, int buf) {
        unsigned int voff = (unsigned int)((((unsigned int)(ks * nTiles + nt0 + ntl) * 32u
                                             + (unsigned int)lane_s) * 16u + (unsigned int)j0) * 2u);
        unsigned int ldsa = (unsigned int)(size_t)(&ldsB[buf][t * 8]);
        asm volatile("global_load_async_to_lds_b128 %0, %1, %2"
                     :: "v"(ldsa), "v"(voff), "s"(B)
                     : "memory");
    };
#else
    auto issueB = [&](int ks, int buf) {
        const v4i* src = (const v4i*)(B + ((size_t)(ks * nTiles + nt0 + ntl) * 32 + lane_s) * 16 + j0);
        *(v4i*)(&ldsB[buf][(size_t)t * 8]) = *src;
    };
#endif

    v8f zero = {0.f, 0.f, 0.f, 0.f, 0.f, 0.f, 0.f, 0.f};
    v8f acc[4];
#pragma unroll
    for (int i = 0; i < 4; ++i) acc[i] = zero;

    issueB(0, 0);                                  // prologue: tile 0 in flight
    for (int ks = 0; ks < ksteps; ++ks) {
        int buf = ks & 1;
#if USE_ASYNC_LDS
        wait_asynccnt0();                          // tile ks landed in LDS
#endif
        __syncthreads();                           // visible block-wide; buf^1 reads done
        if (ks + 1 < ksteps) issueB(ks + 1, buf ^ 1);  // overlap next copy with math

        Frag a;
        if (act) {
            a.u = *(const v16us*)(A + ((size_t)(rowtile * ksteps + ks) * 32 + lane) * 16);
            if (ks + 1 < ksteps)
                __builtin_prefetch(A + ((size_t)(rowtile * ksteps + ks + 1) * 32 + lane) * 16, 0, 1);
        } else {
#pragma unroll
            for (int j = 0; j < 16; ++j) a.u[j] = 0;
        }
#pragma unroll
        for (int nt = 0; nt < 4; ++nt) {
            Frag bfr;
            bfr.u = *(const v16us*)(&ldsB[buf][(nt * 32 + lane) * 16]);
            acc[nt] = __builtin_amdgcn_wmma_f32_16x16x32_bf16(
                false, a.b, false, bfr.b, (short)0, acc[nt], false, false);
        }
    }
    if (!act) return;

    const int ksteps2 = Nf >> 5;
#pragma unroll
    for (int nt = 0; nt < 4; ++nt) {
        int n = (nt0 + nt) * 16 + ln;
        float bv = bias ? bias[n] : 0.f;
#pragma unroll
        for (int r = 0; r < 8; ++r) {
            int m = rowtile * 16 + r + 8 * hi;
            if (m >= rows) continue;
            float v = acc[nt][r] + bv;
            if (RELU) v = fmaxf(v, 0.f);
            if (OUT_MODE == 0) {
                int kk = n & 31, ks2 = n >> 5;
                int hi2, j2;
                if (kk < 8)       { hi2 = 0; j2 = kk; }
                else if (kk < 16) { hi2 = 1; j2 = kk - 8; }
                else if (kk < 24) { hi2 = 0; j2 = kk - 8; }
                else              { hi2 = 1; j2 = kk - 16; }
                unsigned short* op = (unsigned short*)outp;
                op[((size_t)((m >> 4) * ksteps2 + ks2) * 32 + ((m & 15) + (hi2 << 4))) * 16 + j2] = f2bf(v);
            } else {
                ((float*)outp)[(size_t)n * rows + m] = v;   // [E, Q] final layout
            }
        }
    }
}

// ---------------------------------------------------------------------------
// Launcher
// ---------------------------------------------------------------------------
extern "C" void kernel_launch(void* const* d_in, const int* in_sizes, int n_in,
                              void* d_out, int out_size, void* d_ws, size_t ws_size,
                              hipStream_t stream) {
    (void)in_sizes; (void)n_in; (void)out_size; (void)ws_size;
    const float* f0   = (const float*)d_in[0];
    const float* f1   = (const float*)d_in[1];
    const float* f2   = (const float*)d_in[2];
    const float* f3   = (const float*)d_in[3];
    const float* qry  = (const float*)d_in[4];
    const float* qpos = (const float*)d_in[5];
    const float* l2i  = (const float*)d_in[6];
    const float* Wsw  = (const float*)d_in[7];
    const float* bsw  = (const float*)d_in[8];
    const float* pew1 = (const float*)d_in[9];
    const float* peb1 = (const float*)d_in[10];
    const float* pew2 = (const float*)d_in[11];
    const float* peb2 = (const float*)d_in[12];
    const float* cw1  = (const float*)d_in[13];
    const float* cb1  = (const float*)d_in[14];
    const float* cw2  = (const float*)d_in[15];
    const float* cb2  = (const float*)d_in[16];
    const float* cw3  = (const float*)d_in[17];
    const float* cb3  = (const float*)d_in[18];
    const float* cw4  = (const float*)d_in[19];
    const float* cb4  = (const float*)d_in[20];

    char* ws = (char*)d_ws;
    size_t off = 0;
    auto take = [&](size_t bytes) -> char* {
        char* p = ws + off;
        off = (off + bytes + 255) & ~(size_t)255;
        return p;
    };
    unsigned short* featPk = (unsigned short*)take((size_t)40000 * 1024 * 2);
    unsigned short* h1Pk   = (unsigned short*)take((size_t)40000 * 512 * 2);
    unsigned short* h2Pk   = (unsigned short*)take((size_t)40000 * 512 * 2);
    unsigned short* w1Pk   = (unsigned short*)take((size_t)1024 * 512 * 2);
    unsigned short* w2Pk   = (unsigned short*)take((size_t)512 * 512 * 2);
    unsigned short* w3Pk   = (unsigned short*)take((size_t)512 * 512 * 2);
    unsigned short* w4Pk   = (unsigned short*)take((size_t)512 * 128 * 2);
    unsigned short* pw2Pk  = (unsigned short*)take((size_t)256 * 128 * 2);
    unsigned short* h3Pk   = h1Pk;   // h1 dead after GEMM2: reuse for h3

    auto repack = [&](const float* W, unsigned short* Pk, int K, int Nf) {
        int total = (K / 32) * (Nf / 16) * 32;
        repack_w<<<(total + 255) / 256, 256, 0, stream>>>(W, Pk, K, Nf);
    };
    repack(cw1, w1Pk, 1024, 512);
    repack(cw2, w2Pk, 512, 512);
    repack(cw3, w3Pk, 512, 512);
    repack(cw4, w4Pk, 512, 128);
    repack(pew2, pw2Pk, 256, 128);

    sample_kernel<<<2500, 256, 0, stream>>>(f0, f1, f2, f3, qry, qpos, l2i, Wsw, bsw,
                                            pew1, peb1, pw2Pk, peb2, featPk);

    gemm_kernel<0, true ><<<dim3(8, 313), 256, 0, stream>>>(featPk, w1Pk, cb1, h1Pk, 40000, 1024, 512);
    gemm_kernel<0, true ><<<dim3(8, 313), 256, 0, stream>>>(h1Pk,   w2Pk, cb2, h2Pk, 40000, 512,  512);
    gemm_kernel<0, true ><<<dim3(8, 313), 256, 0, stream>>>(h2Pk,   w3Pk, cb3, h3Pk, 40000, 512,  512);
    gemm_kernel<2, false><<<dim3(2, 313), 256, 0, stream>>>(h3Pk,   w4Pk, cb4, d_out, 40000, 512,  128);
}
